// DirResNet2_58523224375718
// MI455X (gfx1250) — compile-verified, matrix operand order checked by reading, not run on hardware
//
#include <hip/hip_runtime.h>

// ---------------------------------------------------------------------------
// DirResNet2 on MI455X (gfx1250): SpMM scatter + BN-folded bf16 WMMA GEMMs.
// Round 2: fused stats+pack pass; async global->LDS weight fill; persistent
// grid-stride GEMM tiles.
// ---------------------------------------------------------------------------

typedef __attribute__((ext_vector_type(16))) __bf16 v16bf;
typedef __attribute__((ext_vector_type(8)))  __bf16 v8bf;
typedef __attribute__((ext_vector_type(8)))  float  v8f;

__device__ __forceinline__ float elu_f(float x) {
    return x > 0.0f ? x : expm1f(x);
}

__device__ __forceinline__ unsigned short f32_to_bf16(float x) {
    unsigned int u = __builtin_bit_cast(unsigned int, x);
    u = (u + 0x7FFFu + ((u >> 16) & 1u)) >> 16;   // round-to-nearest-even
    return (unsigned short)u;
}

// ---------------------------------------------------------------------------
// SpMM: msg[b, row, 0:32] += val * elu(src[b, col, 0:32]); one wave per nnz,
// lane = channel.  src/msg are flat [B, rows4, 32] == [B, rows, 128].
// ---------------------------------------------------------------------------
__global__ void k_spmm(const int* __restrict__ rows, const int* __restrict__ cols,
                       const float* __restrict__ vals, const float* __restrict__ src,
                       float* __restrict__ dst, int nnz,
                       long src_bstride, long dst_bstride, int nbatch) {
    long t = (long)blockIdx.x * blockDim.x + threadIdx.x;
    long e = t >> 5;
    int  c = (int)(t & 31);
    if (e >= nnz) return;
    int   row = rows[e];
    int   col = cols[e];
    float val = vals[e];
    long soff = (long)col * 32 + c;
    long doff = (long)row * 32 + c;
    for (int b = 0; b < nbatch; ++b) {
        float x = elu_f(src[b * src_bstride + soff]);
        atomicAdd(&dst[b * dst_bstride + doff], val * x);
    }
}

// ---------------------------------------------------------------------------
// Fused BN-stats + bf16 A-pack: single streaming pass over the virtual concat
// [elu(base) | msg]. Emits A[r,256] bf16 and per-channel sum / sumsq.
// ---------------------------------------------------------------------------
__global__ void k_stats_pack(const float* __restrict__ base, const float* __restrict__ msg,
                             unsigned short* __restrict__ A,
                             float* __restrict__ sum, float* __restrict__ sumsq,
                             long rowsTotal) {
    int ch = threadIdx.x;            // 0..255
    bool lo = ch < 128;
    int c = ch & 127;
    float s = 0.0f, s2 = 0.0f;
    for (long r = blockIdx.x; r < rowsTotal; r += gridDim.x) {
        float x = lo ? elu_f(base[r * 128 + c]) : msg[r * 128 + c];
        s  += x;
        s2 += x * x;
        A[r * 256 + ch] = f32_to_bf16(x);
    }
    atomicAdd(&sum[ch], s);
    atomicAdd(&sumsq[ch], s2);
}

// ---------------------------------------------------------------------------
// Fold BN affine into weights: W'[n,k] = bf16(w[n,k]*a[k]),
// bias'[n] = b[n] + sum_k c[k]*w[n,k], a = gamma*rsqrt(var+eps), c = beta-mu*a.
// One block, 256 threads.
// ---------------------------------------------------------------------------
__global__ void k_fold(const float* __restrict__ sum, const float* __restrict__ sumsq,
                       const float* __restrict__ gamma, const float* __restrict__ beta,
                       const float* __restrict__ w, const float* __restrict__ b,
                       unsigned short* __restrict__ w2, float* __restrict__ bias2,
                       float invRows) {
    __shared__ float sa[256];
    __shared__ float sc[256];
    int t = threadIdx.x;
    float mu  = sum[t] * invRows;
    float var = sumsq[t] * invRows - mu * mu;
    float a   = gamma[t] * rsqrtf(var + 1e-5f);
    sa[t] = a;
    sc[t] = beta[t] - mu * a;
    __syncthreads();
    float ak = sa[t];
    for (int n = 0; n < 128; ++n)
        w2[n * 256 + t] = f32_to_bf16(w[n * 256 + t] * ak);
    if (t < 128) {
        float acc = b[t];
        for (int k = 0; k < 256; ++k) acc += sc[k] * w[t * 256 + k];
        bias2[t] = acc;
    }
}

// ---------------------------------------------------------------------------
// WMMA GEMM: out[M,128] = A[M,256] @ W'[256,128] (+ bias' [+ resid]).
// Block = 256 threads = 8 waves; wave -> 16x128 strip of a 128-row tile;
// persistent grid-stride over tiles. W' (64KB bf16) is DMA'd into LDS with
// GLOBAL_LOAD_ASYNC_TO_LDS_B128 (ASYNCcnt) — no VGPR staging.
// A-fragment layout per ISA 16-bit 16x32: lanes 0-15 hold K {0..7,16..23},
// lanes 16-31 hold K {8..15,24..31}; B fragment mirrored per-column.
// ---------------------------------------------------------------------------
template <bool RESID>
__global__ __launch_bounds__(256) void k_gemm(const unsigned short* __restrict__ A,
                                              const unsigned short* __restrict__ W2,
                                              const float* __restrict__ bias,
                                              const float* __restrict__ resid,
                                              float* __restrict__ out, long M) {
    __shared__ unsigned short sW[128 * 256];   // 64 KB: W'[n][k]; LDS offset 0

    // Async DMA fill of the folded weights: 256 threads x 16 x b128 = 64 KB.
    // sW is the only static LDS allocation -> it sits at LDS offset 0, so the
    // LDS destination address is just the linear byte offset.
    {
        unsigned           lds0 = (unsigned)threadIdx.x * 16u;
        unsigned long long ga0  = (unsigned long long)W2 + lds0;
#pragma unroll
        for (int i = 0; i < 16; ++i) {
            unsigned           lds = lds0 + (unsigned)i * 4096u;
            unsigned long long ga  = ga0 + (unsigned long long)i * 4096ull;
            asm volatile("global_load_async_to_lds_b128 %0, %1, off"
                         :: "v"(lds), "v"(ga) : "memory");
        }
        asm volatile("s_wait_asynccnt 0x0" ::: "memory");
    }
    __syncthreads();

    const int wave = threadIdx.x >> 5;
    const int lane = threadIdx.x & 31;
    const int mrow = lane & 15;
    const int koff = (lane < 16) ? 0 : 8;

    const __bf16* Ab = (const __bf16*)A;
    const __bf16* Wb = (const __bf16*)sW;

    for (long tile = blockIdx.x; tile * 128 < M; tile += gridDim.x) {
        long m0 = tile * 128 + wave * 16;
        if (m0 >= M) continue;                 // M % 16 == 0; no partial strips

        v8f acc[8];
#pragma unroll
        for (int j = 0; j < 8; ++j) acc[j] = (v8f){0.f, 0.f, 0.f, 0.f, 0.f, 0.f, 0.f, 0.f};

#pragma unroll
        for (int kt = 0; kt < 8; ++kt) {
            const int kbase = kt * 32;
            const __bf16* arow = Ab + (m0 + mrow) * 256 + kbase;
            if (kt < 7) __builtin_prefetch(arow + 32, 0, 1);   // next K-tile of A
            v8bf alo = *(const v8bf*)(arow + koff);
            v8bf ahi = *(const v8bf*)(arow + 16 + koff);
            v16bf af = __builtin_shufflevector(alo, ahi, 0, 1, 2, 3, 4, 5, 6, 7,
                                               8, 9, 10, 11, 12, 13, 14, 15);
#pragma unroll
            for (int j = 0; j < 8; ++j) {
                const __bf16* brow = Wb + (16 * j + mrow) * 256 + kbase;
                v8bf blo = *(const v8bf*)(brow + koff);
                v8bf bhi = *(const v8bf*)(brow + 16 + koff);
                v16bf bf_ = __builtin_shufflevector(blo, bhi, 0, 1, 2, 3, 4, 5, 6, 7,
                                                    8, 9, 10, 11, 12, 13, 14, 15);
                acc[j] = __builtin_amdgcn_wmma_f32_16x16x32_bf16(
                    false, af, false, bf_, (short)0, acc[j], false, false);
            }
        }

        // Epilogue: C/D layout — lane n%16 = column, VGPR r = row r (lanes<16)
        // or r+8 (lanes>=16).
#pragma unroll
        for (int j = 0; j < 8; ++j) {
            const int ncol = 16 * j + mrow;
            const float bb = bias[ncol];
#pragma unroll
            for (int r = 0; r < 8; ++r) {
                long row = m0 + ((lane < 16) ? r : r + 8);
                float val = acc[j][r] + bb;
                if (RESID) val += resid[row * 128 + ncol];
                out[row * 128 + ncol] = val;
            }
        }
    }
}

// ---------------------------------------------------------------------------
// Host-side launcher
// ---------------------------------------------------------------------------
extern "C" void kernel_launch(void* const* d_in, const int* in_sizes, int n_in,
                              void* d_out, int out_size, void* d_ws, size_t ws_size,
                              hipStream_t stream) {
    (void)in_sizes; (void)n_in; (void)out_size; (void)ws_size;

    const long B = 4, N = 30000, F = 60000, C = 128;
    const int  NNZ = 720000;

    const int*   Di_rows  = (const int*)d_in[0];
    const int*   Di_cols  = (const int*)d_in[1];
    const float* Di_vals  = (const float*)d_in[2];
    const int*   DiA_rows = (const int*)d_in[3];
    const int*   DiA_cols = (const int*)d_in[4];
    const float* DiA_vals = (const float*)d_in[5];
    const float* v        = (const float*)d_in[6];
    const float* f        = (const float*)d_in[7];
    const float* bn0_g    = (const float*)d_in[8];
    const float* bn0_b    = (const float*)d_in[9];
    const float* fc0_w    = (const float*)d_in[10];
    const float* fc0_b    = (const float*)d_in[11];
    const float* bn1_g    = (const float*)d_in[12];
    const float* bn1_b    = (const float*)d_in[13];
    const float* fc1_w    = (const float*)d_in[14];
    const float* fc1_b    = (const float*)d_in[15];

    float* v_out = (float*)d_out;               // [B*N, 128]
    float* f_out = (float*)d_out + B * N * C;   // [B*F, 128]

    // Workspace carve-up (16B-aligned slices).
    char* ws = (char*)d_ws;
    float* msg1 = (float*)ws;            ws += (size_t)B * F * 128 * sizeof(float);
    float* msg2 = (float*)ws;            ws += (size_t)B * N * 128 * sizeof(float);
    unsigned short* A1 = (unsigned short*)ws;  ws += (size_t)B * F * 256 * sizeof(unsigned short);
    unsigned short* A2 = (unsigned short*)ws;  ws += (size_t)B * N * 256 * sizeof(unsigned short);
    unsigned short* W2a = (unsigned short*)ws; ws += 128 * 256 * sizeof(unsigned short);
    unsigned short* W2b = (unsigned short*)ws; ws += 128 * 256 * sizeof(unsigned short);
    float* bias1 = (float*)ws;           ws += 128 * sizeof(float);
    float* bias2 = (float*)ws;           ws += 128 * sizeof(float);
    float* ssum  = (float*)ws;           ws += 256 * sizeof(float);
    float* ssq   = (float*)ws;           ws += 256 * sizeof(float);

    const long rowsF = B * F;            // 240000
    const long rowsV = B * N;            // 120000

    // ---- stage 1: vertex -> face ------------------------------------------
    hipMemsetAsync(msg1, 0, (size_t)B * F * 128 * sizeof(float), stream);
    hipMemsetAsync(ssum, 0, 256 * sizeof(float), stream);
    hipMemsetAsync(ssq,  0, 256 * sizeof(float), stream);

    {
        long threads = (long)NNZ * 32;
        k_spmm<<<dim3((unsigned)((threads + 255) / 256)), dim3(256), 0, stream>>>(
            Di_rows, Di_cols, Di_vals, v, msg1, NNZ, N * 128, F * 128, (int)B);
    }
    k_stats_pack<<<dim3(4096), dim3(256), 0, stream>>>(f, msg1, A1, ssum, ssq, rowsF);
    k_fold<<<dim3(1), dim3(256), 0, stream>>>(ssum, ssq, bn0_g, bn0_b, fc0_w, fc0_b,
                                              W2a, bias1, 1.0f / (float)rowsF);
    k_gemm<false><<<dim3(960), dim3(256), 0, stream>>>(
        A1, W2a, bias1, nullptr, f_out, rowsF);

    // ---- stage 2: face -> vertex ------------------------------------------
    hipMemsetAsync(msg2, 0, (size_t)B * N * 128 * sizeof(float), stream);
    hipMemsetAsync(ssum, 0, 256 * sizeof(float), stream);
    hipMemsetAsync(ssq,  0, 256 * sizeof(float), stream);

    {
        long threads = (long)NNZ * 32;
        k_spmm<<<dim3((unsigned)((threads + 255) / 256)), dim3(256), 0, stream>>>(
            DiA_rows, DiA_cols, DiA_vals, f_out, msg2, NNZ, F * 128, N * 128, (int)B);
    }
    k_stats_pack<<<dim3(4096), dim3(256), 0, stream>>>(v, msg2, A2, ssum, ssq, rowsV);
    k_fold<<<dim3(1), dim3(256), 0, stream>>>(ssum, ssq, bn1_g, bn1_b, fc1_w, fc1_b,
                                              W2b, bias2, 1.0f / (float)rowsV);
    k_gemm<true><<<dim3(960), dim3(256), 0, stream>>>(
        A2, W2b, bias2, v, v_out, rowsV);
}
